// SelfAttentionEasy_63960652972223
// MI455X (gfx1250) — compile-verified, matrix operand order checked by reading, not run on hardware
//
#include <hip/hip_runtime.h>
#include <hip/hip_bf16.h>

// CDNA5 / gfx1250 flash-attention-style kernel (wave32, v_wmma_f32_16x16x32_f16)
// Phase 1: bandwidth-bound pre-pass packs K and V into f16 WMMA-B-operand layout.
// Phase 2: main kernel, 32 Q-rows (two 16-row tiles) per wave so every K/V
//          operand register feeds two WMMAs; row sums via P x ones WMMA.

typedef __attribute__((ext_vector_type(16))) _Float16 v16h;
typedef __attribute__((ext_vector_type(8)))  float    v8f;

#define S_LEN  2048
#define D_DIM  64
#define NHEAD  32              // b*h = 2*16
#define WAVES  4               // waves per block in main kernel
#define RT     2               // 16-row Q tiles per wave
#define NTILES (S_LEN / 16)    // 128 key tiles of 16
#define NCHUNK (S_LEN / 32)    // 64  key chunks of 32

// Base K-index of the contiguous element pair held in "VGPR v" of a 16-bit
// A/B WMMA operand (ISA 7.12.2): lanes 0-15 (half=0): V0..V3 -> K=0..7,
// V4..V7 -> K=16..23 ; lanes 16-31 (half=1): +8.
__device__ __forceinline__ int kidx_pair(int v, int half) {
    return 2 * v + 8 * half + ((v >= 4) ? 8 : 0);
}

// ---------------------------------------------------------------------------
// Pre-pass 1: pack K^T (per head [d=64, s] f32) into B-operand f16 fragments.
// Layout: kp[((head*NTILES + nt)*32 + lane)*32 + e]; 64 B contiguous per lane.
// ---------------------------------------------------------------------------
__global__ __launch_bounds__(32)
void pack_k_kernel(const float* __restrict__ K, _Float16* __restrict__ kp)
{
    const int lane = threadIdx.x & 31;
    const int half = lane >> 4;
    const int l16  = lane & 15;
    const int nt   = blockIdx.x;     // key tile of 16
    const int bh   = blockIdx.y;

    const float* k = K + (size_t)bh * D_DIM * S_LEN;
    const int col  = nt * 16 + l16;

    v16h b0, b1;
    #pragma unroll
    for (int vv = 0; vv < 8; ++vv) {
        const int kk = kidx_pair(vv, half);          // d index
        b0[2 * vv + 0] = (_Float16)k[(size_t)(kk +  0) * S_LEN + col];
        b0[2 * vv + 1] = (_Float16)k[(size_t)(kk +  1) * S_LEN + col];
        b1[2 * vv + 0] = (_Float16)k[(size_t)(kk + 32) * S_LEN + col];
        b1[2 * vv + 1] = (_Float16)k[(size_t)(kk + 33) * S_LEN + col];
    }
    _Float16* dst = kp + ((size_t)(bh * NTILES + nt) * 32 + lane) * 32;
    *(v16h*)(dst)      = b0;
    *(v16h*)(dst + 16) = b1;
}

// ---------------------------------------------------------------------------
// Pre-pass 2: pack V (per head [s, d=64] f32) into B-operand f16 fragments
// for the P*V GEMM (K-dim = 32 keys, N = 16 d-columns).
// Layout: vp[(((head*NCHUNK + c)*4 + dt)*32 + lane)*16 + e]; 32 B per lane.
// ---------------------------------------------------------------------------
__global__ __launch_bounds__(32)
void pack_v_kernel(const float* __restrict__ V, _Float16* __restrict__ vp)
{
    const int lane = threadIdx.x & 31;
    const int half = lane >> 4;
    const int l16  = lane & 15;
    const int dt   = blockIdx.x & 3;     // d-column tile of 16
    const int c    = blockIdx.x >> 2;    // key chunk of 32
    const int bh   = blockIdx.y;

    const float* v = V + (size_t)bh * S_LEN * D_DIM;

    v16h vb;
    #pragma unroll
    for (int vv = 0; vv < 8; ++vv) {
        const int kk = kidx_pair(vv, half);          // key index within chunk
        const float* src = v + (size_t)(c * 32 + kk) * D_DIM + dt * 16 + l16;
        vb[2 * vv + 0] = (_Float16)src[0];
        vb[2 * vv + 1] = (_Float16)src[D_DIM];
    }
    _Float16* dst = vp + ((size_t)((bh * NCHUNK + c) * 4 + dt) * 32 + lane) * 16;
    *(v16h*)dst = vb;
}

// ---------------------------------------------------------------------------
// Main kernel: one wave32 per 32 Q rows (two 16-row tiles), flash loop on keys.
// ---------------------------------------------------------------------------
__global__ __launch_bounds__(WAVES * 32)
void attn_fwd_wmma(const float* __restrict__ Q,
                   const _Float16* __restrict__ kp,
                   const _Float16* __restrict__ vp,
                   float* __restrict__ O)
{
    // wave-private scratch for the P (16x32 f32) C->A layout transposes
    __shared__ __align__(16) float pbuf[WAVES][RT][16 * 32];

    const int lane = threadIdx.x & 31;
    const int wave = threadIdx.x >> 5;
    const int half = lane >> 4;
    const int l16  = lane & 15;

    const int bh   = blockIdx.y;                     // 0..31  (b*h)
    const int tile = blockIdx.x * WAVES + wave;      // 0..63  (32-row tiles)
    const int row0 = tile * (16 * RT);

    const float* q = Q + (size_t)bh * S_LEN * D_DIM;
    float*       o = O + (size_t)bh * S_LEN * D_DIM;

    // Fold (1/sqrt(d)) * log2(e) into Q so the S tile lands in exp2 domain.
    const float qscale = 0.125f * 1.44269504088896340736f;

    // ---- Load the 2x(16x64) Q tiles once, scale, f32 -> f16 A operands ----
    v16h qa[RT][2];                                  // [row tile][K = 0..31 / 32..63]
    #pragma unroll
    for (int rt = 0; rt < RT; ++rt) {
        const float* qrow = q + (size_t)(row0 + rt * 16 + l16) * D_DIM;
        #pragma unroll
        for (int vv = 0; vv < 8; ++vv) {
            const int kk = kidx_pair(vv, half);
            const float2 x0 = *(const float2*)(qrow + kk);
            const float2 x1 = *(const float2*)(qrow + kk + 32);
            qa[rt][0][2 * vv + 0] = (_Float16)(x0.x * qscale);
            qa[rt][0][2 * vv + 1] = (_Float16)(x0.y * qscale);
            qa[rt][1][2 * vv + 0] = (_Float16)(x1.x * qscale);
            qa[rt][1][2 * vv + 1] = (_Float16)(x1.y * qscale);
        }
    }

    // All-ones B operand: P x ones puts the f16-P row sums in C layout,
    // replicated across every lane of the row group -> no shuffle reduction.
    v16h ones;
    #pragma unroll
    for (int e = 0; e < 16; ++e) ones[e] = (_Float16)1.0f;

    v8f oacc[RT][4];              // O tiles, 4 d-column tiles of 16, C/D layout
    v8f lacc[RT];                 // row-sum accumulators, C/D layout
    #pragma unroll
    for (int rt = 0; rt < RT; ++rt) {
        #pragma unroll
        for (int r = 0; r < 8; ++r) {
            lacc[rt][r] = 0.0f;
            oacc[rt][0][r] = 0.0f; oacc[rt][1][r] = 0.0f;
            oacc[rt][2][r] = 0.0f; oacc[rt][3][r] = 0.0f;
        }
    }

    // per-lane packed-operand base pointers (advance by constant strides)
    const _Float16* kbase = kp + ((size_t)(bh * NTILES) * 32 + lane) * 32;
    const _Float16* vbase = vp + ((size_t)(bh * NCHUNK) * 4 * 32 + lane) * 16;

    for (int c = 0; c < NCHUNK; ++c) {
        const _Float16* kc = kbase + (size_t)(2 * c) * (32 * 32); // 2 tiles/chunk
        const _Float16* vc = vbase + (size_t)c * (4 * 32 * 16);

        if (c + 1 < NCHUNK) {  // pull next chunk's fragments toward the WGP
            __builtin_prefetch(kc + 2 * (32 * 32), 0, 3);
            __builtin_prefetch(vc + 4 * 32 * 16, 0, 3);
        }

        // ---- S = Q K^T: each K operand pair feeds both row tiles ----
        #pragma unroll
        for (int nb = 0; nb < 2; ++nb) {
            const _Float16* kt = kc + (size_t)nb * (32 * 32);
            const v16h b0 = *(const v16h*)(kt);        // d = 0..31
            const v16h b1 = *(const v16h*)(kt + 16);   // d = 32..63
            #pragma unroll
            for (int rt = 0; rt < RT; ++rt) {
                v8f acc = {};
                acc = __builtin_amdgcn_wmma_f32_16x16x32_f16(false, qa[rt][0], false, b0,
                                                             (short)0, acc, false, false);
                acc = __builtin_amdgcn_wmma_f32_16x16x32_f16(false, qa[rt][1], false, b1,
                                                             (short)0, acc, false, false);
                // p = 2^s via raw v_exp_f32; stash C-layout tile to LDS
                #pragma unroll
                for (int r = 0; r < 8; ++r) {
                    const float p = __builtin_amdgcn_exp2f(acc[r]);
                    pbuf[wave][rt][(r + 8 * half) * 32 + nb * 16 + l16] = p;
                }
            }
        }

        // DS pipeline is in-order per wave; explicit wait before cross-lane reads
        asm volatile("s_wait_dscnt 0x0" ::: "memory");

        // ---- reload P tiles (16x32) from LDS in A-operand f16 layout ----
        v16h pa[RT];
        #pragma unroll
        for (int rt = 0; rt < RT; ++rt) {
            #pragma unroll
            for (int vv = 0; vv < 8; ++vv) {
                const int kk = kidx_pair(vv, half);    // key index within chunk
                const float2 x = *(const float2*)&pbuf[wave][rt][l16 * 32 + kk];
                pa[rt][2 * vv + 0] = (_Float16)x.x;
                pa[rt][2 * vv + 1] = (_Float16)x.y;
            }
        }

        // ---- O += P * V : each V operand feeds both row tiles ----
        #pragma unroll
        for (int dt = 0; dt < 4; ++dt) {
            const v16h vb = *(const v16h*)(vc + (size_t)dt * (32 * 16));
            #pragma unroll
            for (int rt = 0; rt < RT; ++rt)
                oacc[rt][dt] = __builtin_amdgcn_wmma_f32_16x16x32_f16(false, pa[rt], false, vb,
                                                                      (short)0, oacc[rt][dt],
                                                                      false, false);
        }

        // ---- row sums on the matrix pipe: lacc += P * ones ----
        #pragma unroll
        for (int rt = 0; rt < RT; ++rt)
            lacc[rt] = __builtin_amdgcn_wmma_f32_16x16x32_f16(false, pa[rt], false, ones,
                                                              (short)0, lacc[rt],
                                                              false, false);
    }

    // ---- normalize (rcp + mul) and store: (m = r + 8*half, n = l16) ----
    #pragma unroll
    for (int rt = 0; rt < RT; ++rt) {
        #pragma unroll
        for (int dt = 0; dt < 4; ++dt) {
            #pragma unroll
            for (int r = 0; r < 8; ++r) {
                const int m = rt * 16 + r + 8 * half;
                const float inv = __builtin_amdgcn_rcpf(lacc[rt][r]);
                o[(size_t)(row0 + m) * D_DIM + dt * 16 + l16] = oacc[rt][dt][r] * inv;
            }
        }
    }
}

extern "C" void kernel_launch(void* const* d_in, const int* in_sizes, int n_in,
                              void* d_out, int out_size, void* d_ws, size_t ws_size,
                              hipStream_t stream) {
    (void)in_sizes; (void)n_in; (void)out_size; (void)ws_size;
    const float* q = (const float*)d_in[0];
    const float* k = (const float*)d_in[1];
    const float* v = (const float*)d_in[2];
    float* out = (float*)d_out;

    // workspace: packed-f16 K then packed-f16 V (8.39 MB each)
    const size_t kpack_elems = (size_t)NHEAD * NTILES * 32 * 32;      // 4.19M halves
    _Float16* kp = (_Float16*)d_ws;
    _Float16* vp = kp + kpack_elems;

    // Phase 1: pack K and V (independent; stream-ordered before main kernel)
    pack_k_kernel<<<dim3(NTILES, NHEAD), 32, 0, stream>>>(k, kp);
    pack_v_kernel<<<dim3(NCHUNK * 4, NHEAD), 32, 0, stream>>>(v, vp);

    // Phase 2: attention, 32 Q rows per wave
    dim3 grid(NTILES / RT / WAVES, NHEAD);   // 16 x 32 blocks
    dim3 block(WAVES * 32);                  // 4 wave32 per block
    attn_fwd_wmma<<<grid, block, 0, stream>>>(q, kp, vp, out);
}
